// TriangleMultiplication_63763084476750
// MI455X (gfx1250) — matrix operand order repaired and checked by reading, not compiled, not run on hardware
//
#include <hip/hip_runtime.h>
#include <hip/hip_bf16.h>

typedef __attribute__((ext_vector_type(16))) _Float16 v16h;
typedef __attribute__((ext_vector_type(8)))  _Float16 v8h;
typedef __attribute__((ext_vector_type(8)))  float    v8f;
typedef __attribute__((ext_vector_type(4)))  float    v4f;

#define NN 512
#define CZ 128
#define RTOT (NN * NN)          // 262144 flattened (i,k)/(i,j) rows
#define WMAT 16384              // 128*128 elements per weight matrix

static __device__ __forceinline__ v16h hcat(v8h lo, v8h hi) {
    return __builtin_shufflevector(lo, hi, 0,1,2,3,4,5,6,7,8,9,10,11,12,13,14,15);
}

static __device__ __forceinline__ v8f wmma_f16(v16h a, v16h b, v8f c) {
    return __builtin_amdgcn_wmma_f32_16x16x32_f16(false, a, false, b, (short)0, c, false, false);
}

static __device__ __forceinline__ float sigmoidf(float x) {
    return 1.0f / (1.0f + __expf(-x));
}

// B-fragment: lane l holds column (base + l%16), K chunks {k0+sel..+7} and {k0+sel+16..+23};
// caller passes pointer already offset by n*ldb + k0 + ((lane<16)?0:8).
static __device__ __forceinline__ v16h loadB(const _Float16* p) {
    return hcat(*(const v8h*)(p), *(const v8h*)(p + 16));
}

// A-fragment: lane l holds row (l%16), K chunks {k0+sel..+7} and {k0+sel+16..+23},
// sel = 0 (lanes 0-15) or 8 (lanes 16-31). Pointer already offset to row + k0 + sel.
static __device__ __forceinline__ v16h loadA(const _Float16* p) {
    return hcat(*(const v8h*)(p), *(const v8h*)(p + 16));
}

// ---------------------------------------------------------------------------
// Kernel 1: convert + transpose weights  f32 [in][out]  ->  f16 [out][in]
// ---------------------------------------------------------------------------
__global__ void prep_weights(const float* __restrict__ wap, const float* __restrict__ wag,
                             const float* __restrict__ wbp, const float* __restrict__ wbg,
                             const float* __restrict__ wg,  const float* __restrict__ wz,
                             _Float16* __restrict__ Wt) {
    int mat = blockIdx.x >> 6;
    int e   = ((blockIdx.x & 63) << 8) | threadIdx.x;   // 0..16383
    const float* W;
    switch (mat) {
        case 0: W = wap; break;
        case 1: W = wag; break;
        case 2: W = wbp; break;
        case 3: W = wbg; break;
        case 4: W = wg;  break;
        default: W = wz; break;
    }
    int i = e >> 7;        // input index (K)
    int o = e & 127;       // output index (N)
    Wt[(size_t)mat * WMAT + (size_t)o * 128 + i] = (_Float16)W[e];
}

// ---------------------------------------------------------------------------
// Kernel 2: layernorm(z) + 5 projections + gating. Block = 256 threads (8
// waves) handles 16 rows; wave w owns n-tile w for all three outputs.
// Outputs channel-major f16: a_t[c][r], b_t[c][r], g_t[c][r].
// ---------------------------------------------------------------------------
__global__ void __launch_bounds__(256)
proj_kernel(const float* __restrict__ z,
            const float* __restrict__ ln_in_w, const float* __restrict__ ln_in_b,
            const _Float16* __restrict__ Wt,
            const float* __restrict__ b_ap, const float* __restrict__ b_ag,
            const float* __restrict__ b_bp, const float* __restrict__ b_bg,
            const float* __restrict__ b_g,
            _Float16* __restrict__ a_t, _Float16* __restrict__ b_t,
            _Float16* __restrict__ g_t) {
    __shared__ _Float16 lnz[16 * 136];   // 16 rows, padded stride (16B aligned rows)

    const int t    = threadIdx.x;
    const int lane = t & 31;
    const int wv   = t >> 5;
    const int r0   = blockIdx.x * 16;

    // ---- layernorm: 16 threads per row, 8 elems per thread ----
    {
        const int row  = t >> 4;
        const int col8 = (t & 15) * 8;
        const float* zr = z + ((size_t)(r0 + row)) * CZ + col8;
        v4f x0 = *(const v4f*)(zr);
        v4f x1 = *(const v4f*)(zr + 4);
        float s = 0.f, q = 0.f;
#pragma unroll
        for (int i = 0; i < 4; i++) { s += x0[i] + x1[i]; q += x0[i]*x0[i] + x1[i]*x1[i]; }
#pragma unroll
        for (int off = 8; off; off >>= 1) {
            s += __shfl_xor(s, off, 32);
            q += __shfl_xor(q, off, 32);
        }
        float mean = s * (1.0f / 128.0f);
        float var  = q * (1.0f / 128.0f) - mean * mean;
        float rstd = rsqrtf(var + 1e-5f);
        v4f w0 = *(const v4f*)(ln_in_w + col8);
        v4f w1 = *(const v4f*)(ln_in_w + col8 + 4);
        v4f c0 = *(const v4f*)(ln_in_b + col8);
        v4f c1 = *(const v4f*)(ln_in_b + col8 + 4);
        v8h h;
#pragma unroll
        for (int i = 0; i < 4; i++) {
            h[i]     = (_Float16)((x0[i] - mean) * rstd * w0[i] + c0[i]);
            h[i + 4] = (_Float16)((x1[i] - mean) * rstd * w1[i] + c1[i]);
        }
        *(v8h*)(lnz + row * 136 + col8) = h;
    }
    __syncthreads();

    const int m    = lane & 15;
    const int selA = (lane < 16) ? 0 : 8;
    const int selB = (lane < 16) ? 0 : 8;
    const int n    = wv * 16 + m;              // this wave's output column
    const size_t rb = (size_t)r0 + ((lane >> 4) * 8);

    // hoist all 4 K-step A-fragments (zn rows) from LDS once; reused by 5 GEMMs
    v16h A[4];
#pragma unroll
    for (int ks = 0; ks < 4; ks++)
        A[ks] = loadA(lnz + m * 136 + ks * 32 + selA);

    // gated pair helper: sigmoid(zn@Wg + bg) * (zn@Wp + bp) -> dst[c][r]
    auto gated_pair = [&](const _Float16* Wp, const _Float16* Wg,
                          const float* bp, const float* bg, _Float16* dst) {
        v8f accP, accG;
        {
            float pv = bp[n], gv = bg[n];
#pragma unroll
            for (int i = 0; i < 8; i++) { accP[i] = pv; accG[i] = gv; }
        }
#pragma unroll
        for (int ks = 0; ks < 4; ks++) {
            const size_t off = (size_t)n * 128 + ks * 32 + selB;
            v16h Bp = loadB(Wp + off);
            v16h Bg = loadB(Wg + off);
            accP = wmma_f16(A[ks], Bp, accP);
            accG = wmma_f16(A[ks], Bg, accG);
        }
        v8h h;
#pragma unroll
        for (int v = 0; v < 8; v++)
            h[v] = (_Float16)(sigmoidf(accG[v]) * accP[v]);
        *(v8h*)(dst + (size_t)n * RTOT + rb) = h;   // packed b128: rows rb..rb+7, col n
    };

    gated_pair(Wt,            Wt + WMAT,     b_ap, b_ag, a_t);
    gated_pair(Wt + 2 * WMAT, Wt + 3 * WMAT, b_bp, b_bg, b_t);

    // g = sigmoid(zn @ w_g + b_g)
    {
        v8f acc;
        { float gv = b_g[n];
#pragma unroll
          for (int i = 0; i < 8; i++) acc[i] = gv; }
        const _Float16* Wg = Wt + 4 * WMAT;
#pragma unroll
        for (int ks = 0; ks < 4; ks++)
            acc = wmma_f16(A[ks], loadB(Wg + (size_t)n * 128 + ks * 32 + selB), acc);
        v8h h;
#pragma unroll
        for (int v = 0; v < 8; v++) h[v] = (_Float16)sigmoidf(acc[v]);
        *(v8h*)(g_t + (size_t)n * RTOT + rb) = h;
    }
}

// ---------------------------------------------------------------------------
// Kernel 3: fused triangle einsum + layernorm(c) + out-proj + gate.
// Block = 256 threads (8 waves) computes one 32x32 (i,j) tile, all 128 c.
// Each wave: 16 channels, 2x2 WMMA tiles per channel (4 WMMAs per 4 loads
// per K-step -> 2x the flop/byte of a 16x16 tile). t tile staged in LDS as
// f16 (1024 pos x 136-half padded rows = 278.5 KB of the 320 KB WGP LDS).
// ---------------------------------------------------------------------------
__global__ void __launch_bounds__(256)
tri_fused(const _Float16* __restrict__ a_t, const _Float16* __restrict__ b_t,
          const _Float16* __restrict__ g_t, const _Float16* __restrict__ Wz,
          const float* __restrict__ ln_out_w, const float* __restrict__ ln_out_b,
          const float* __restrict__ b_z, float* __restrict__ out) {
    __shared__ _Float16 tbuf[1024 * 136];    // t tile, f16, padded row stride
    __shared__ float s_lw[128], s_lb[128];

    const int t    = threadIdx.x;
    const int lane = t & 31;
    const int wv   = t >> 5;
    const int ti   = blockIdx.x >> 4;
    const int tj   = blockIdx.x & 15;
    const int i0   = ti * 32, j0 = tj * 32;

    if (t < 128) { s_lw[t] = ln_out_w[t]; s_lb[t] = ln_out_b[t]; }

    const int m    = lane & 15;
    const int selA = (lane < 16) ? 0 : 8;
    const int selB = (lane < 16) ? 0 : 8;
    const int Mb   = (lane >> 4) * 8;

    // ---- phase 1: per-channel 32x32x512 contraction (2x2 WMMA tiles),
    //      16 channels per wave, software-pipelined K loop ----
#pragma unroll 1
    for (int cc = 0; cc < 16; cc++) {
        const int c = wv * 16 + cc;
        const _Float16* Ab0 = a_t + (size_t)c * RTOT + (size_t)(i0 + m) * NN + selA;
        const _Float16* Ab1 = Ab0 + 16 * NN;
        const _Float16* Bb0 = b_t + (size_t)c * RTOT + (size_t)(j0 + m) * NN + selB;
        const _Float16* Bb1 = Bb0 + 16 * NN;
        v8f acc00 = {}, acc01 = {}, acc10 = {}, acc11 = {};
        v16h A0 = loadA(Ab0), A1 = loadA(Ab1);
        v16h B0 = loadB(Bb0), B1 = loadB(Bb1);
#pragma unroll 1
        for (int k0 = 32; k0 < NN; k0 += 32) {
            v16h A0n = loadA(Ab0 + k0);
            v16h A1n = loadA(Ab1 + k0);
            v16h B0n = loadB(Bb0 + k0);
            v16h B1n = loadB(Bb1 + k0);
            acc00 = wmma_f16(A0, B0, acc00);
            acc01 = wmma_f16(A0, B1, acc01);
            acc10 = wmma_f16(A1, B0, acc10);
            acc11 = wmma_f16(A1, B1, acc11);
            A0 = A0n; A1 = A1n; B0 = B0n; B1 = B1n;
        }
        acc00 = wmma_f16(A0, B0, acc00);
        acc01 = wmma_f16(A0, B1, acc01);
        acc10 = wmma_f16(A1, B0, acc10);
        acc11 = wmma_f16(A1, B1, acc11);

        // store 4 tiles to LDS as f16: pos = i_local*32 + j_local
#pragma unroll
        for (int v = 0; v < 8; v++) {
            tbuf[(size_t)((Mb + v)      * 32 + m)      * 136 + c] = (_Float16)acc00[v];
            tbuf[(size_t)((Mb + v)      * 32 + m + 16) * 136 + c] = (_Float16)acc01[v];
            tbuf[(size_t)((Mb + v + 16) * 32 + m)      * 136 + c] = (_Float16)acc10[v];
            tbuf[(size_t)((Mb + v + 16) * 32 + m + 16) * 136 + c] = (_Float16)acc11[v];
        }
    }
    __syncthreads();

    // ---- phase 2: layernorm over c, 4 rows per thread, in-place f16 rewrite ----
#pragma unroll 1
    for (int p = 0; p < 4; p++) {
        _Float16* row = tbuf + (size_t)(t + p * 256) * 136;
        float s = 0.f, q = 0.f;
#pragma unroll
        for (int c = 0; c < 128; c += 8) {
            v8h x = *(const v8h*)(row + c);
#pragma unroll
            for (int i = 0; i < 8; i++) { float f = (float)x[i]; s += f; q += f * f; }
        }
        float mean = s * (1.0f / 128.0f);
        float var  = q * (1.0f / 128.0f) - mean * mean;
        float rstd = rsqrtf(var + 1e-5f);
#pragma unroll
        for (int c = 0; c < 128; c += 8) {
            v8h x = *(const v8h*)(row + c);
            v8h h;
#pragma unroll
            for (int i = 0; i < 8; i++)
                h[i] = (_Float16)(((float)x[i] - mean) * rstd * s_lw[c + i] + s_lb[c + i]);
            *(v8h*)(row + c) = h;
        }
    }
    __syncthreads();

    // ---- phase 3: [1024 x 128] @ Wz + b_z, gate with sigmoid(g), store f32.
    //      8 M-tiles per wave; A-fragments hoisted from LDS, reused by 8 N-tiles ----
#pragma unroll 1
    for (int mt = wv * 8; mt < wv * 8 + 8; mt++) {
        const _Float16* Arow = tbuf + (size_t)(mt * 16 + m) * 136 + selA;
        v16h A[4];
#pragma unroll
        for (int ks = 0; ks < 4; ks++)
            A[ks] = loadA(Arow + ks * 32);

        // lane's 8 consecutive output rows for this M-tile:
        // pos = mt*16 + Mb + v  ->  i_local = mt>>1, j_local = (mt&1)*16 + Mb + v
        const size_t rbase = (size_t)(i0 + (mt >> 1)) * NN + j0 + (mt & 1) * 16 + Mb;

#pragma unroll 1
        for (int nt = 0; nt < 8; nt++) {
            const int nn = nt * 16 + m;           // output channel cz / Wz row
            v8f acc;
            { float bv = b_z[nn];
#pragma unroll
              for (int i = 0; i < 8; i++) acc[i] = bv; }
            const _Float16* Bz = Wz + (size_t)nn * 128 + selB;
#pragma unroll
            for (int ks = 0; ks < 4; ks++)
                acc = wmma_f16(A[ks], loadB(Bz + ks * 32), acc);
            v8h gv = *(const v8h*)(g_t + (size_t)nn * RTOT + rbase);
#pragma unroll
            for (int v = 0; v < 8; v++)
                out[(rbase + v) * CZ + nn] = (float)gv[v] * acc[v];
        }
    }
}

// ---------------------------------------------------------------------------
extern "C" void kernel_launch(void* const* d_in, const int* in_sizes, int n_in,
                              void* d_out, int out_size, void* d_ws, size_t ws_size,
                              hipStream_t stream) {
    const float* z        = (const float*)d_in[0];
    const float* ln_in_w  = (const float*)d_in[1];
    const float* ln_in_b  = (const float*)d_in[2];
    const float* ln_out_w = (const float*)d_in[3];
    const float* ln_out_b = (const float*)d_in[4];
    const float* w_ap = (const float*)d_in[5];  const float* b_ap = (const float*)d_in[6];
    const float* w_ag = (const float*)d_in[7];  const float* b_ag = (const float*)d_in[8];
    const float* w_bp = (const float*)d_in[9];  const float* b_bp = (const float*)d_in[10];
    const float* w_bg = (const float*)d_in[11]; const float* b_bg = (const float*)d_in[12];
    const float* w_g  = (const float*)d_in[13]; const float* b_g  = (const float*)d_in[14];
    const float* w_z  = (const float*)d_in[15]; const float* b_z  = (const float*)d_in[16];
    float* out = (float*)d_out;

    // workspace carve-up (f16): 6 weights (192KB) + a_t/b_t/g_t (64MB each)
    char* ws = (char*)d_ws;
    _Float16* Wt  = (_Float16*)ws;
    _Float16* a_t = (_Float16*)(ws + 6 * WMAT * sizeof(_Float16));
    _Float16* b_t = a_t + (size_t)CZ * RTOT;
    _Float16* g_t = b_t + (size_t)CZ * RTOT;

    prep_weights<<<6 * 64, 256, 0, stream>>>(w_ap, w_ag, w_bp, w_bg, w_g, w_z, Wt);

    proj_kernel<<<RTOT / 16, 256, 0, stream>>>(z, ln_in_w, ln_in_b, Wt,
                                               b_ap, b_ag, b_bp, b_bg, b_g,
                                               a_t, b_t, g_t);

    tri_fused<<<(NN / 32) * (NN / 32), 256, 0, stream>>>(a_t, b_t, g_t,
                                                         Wt + 5 * WMAT,
                                                         ln_out_w, ln_out_b, b_z, out);
}
